// AirfoilGNN_70239895159415
// MI455X (gfx1250) — compile-verified, matrix operand order checked by reading, not run on hardware
//
#include <hip/hip_runtime.h>
#include <cstddef>

// ---------------------------------------------------------------------------
// AirfoilGNN forward on gfx1250 (MI455X): all Linear layers via
// v_wmma_f32_16x16x32_f16 (f16 in, f32 accumulate), BN folded at prep time.
// M=32 rows per block: each wave runs two independent WMMA accumulator
// chains sharing one B-fragment load (better XDL pipelining + B reuse).
// ---------------------------------------------------------------------------

typedef _Float16 half_t;
typedef __attribute__((ext_vector_type(16))) _Float16 v16h;
typedef __attribute__((ext_vector_type(8)))  _Float16 v8h;
typedef __attribute__((ext_vector_type(8)))  float    v8f;

#define N_NODES 100000
#define N_EDGES 600000
#define BN_EPS  1e-5f

// ===========================================================================
// Weight prep: fold eval-BN into Linear, convert to f16, pack into the WMMA
// B-fragment tile layout: tile t = nt*(Kp/32)+kt ; within tile 512 halves:
//   half index = lane*16 + j ; lane<16: n=nt*16+lane, j<8 -> k=kt*32+j,
//   j>=8 -> k=kt*32+j+8 ; lane>=16: same with k += 8.
// ===========================================================================
__global__ void prep_weights_kernel(const float* __restrict__ W,
                                    const float* __restrict__ B,
                                    const float* __restrict__ g,
                                    const float* __restrict__ be,
                                    const float* __restrict__ rm,
                                    const float* __restrict__ rv,
                                    int K, int N, int Kp, int Np,
                                    half_t* __restrict__ wout,
                                    float* __restrict__ bout)
{
    int tid = blockIdx.x * 256 + threadIdx.x;
    // packed bias (BN-folded), zero padded to Np
    if (tid < Np) {
        float bb = 0.0f;
        if (tid < N) {
            bb = B[tid];
            if (g) {
                float s = g[tid] * rsqrtf(rv[tid] + BN_EPS);
                bb = (bb - rm[tid]) * s + be[tid];
            }
        }
        bout[tid] = bb;
    }
    int total = Kp * Np;
    if (tid >= total) return;
    int tile = tid >> 9, idx = tid & 511;
    int lane = idx >> 4, j = idx & 15;
    int kch = Kp >> 5;
    int kt = tile % kch, nt = tile / kch;
    int k = kt * 32 + ((j < 8) ? j : (j + 8)) + ((lane & 16) ? 8 : 0);
    int n = nt * 16 + (lane & 15);
    float v = 0.0f;
    if (k < K && n < N) {
        v = W[(size_t)k * N + n];
        if (g) v *= g[n] * rsqrtf(rv[n] + BN_EPS);
    }
    wout[tid] = (half_t)v;
}

// ===========================================================================
// WMMA layer primitives. Block = 8 wave32 (256 threads), M = 32 rows.
// xb/yb are LDS row-major f16 buffers; strides multiple of 8 halves so
// every v8h access is 16B-aligned ds_load_b128 / two WMMAs share one B load.
// ===========================================================================
template<int KP, int NP, bool RELU>
__device__ __forceinline__
void layer_lds(const half_t* __restrict__ wp, const float* __restrict__ bp,
               const half_t* __restrict__ xb, int sx,
               half_t* __restrict__ yb, int sy,
               int lane, int wave)
{
    constexpr int KCH = KP / 32;
    constexpr int NT  = NP / 16;
    const int c15  = lane & 15;
    const int koff = (lane & 16) ? 8 : 0;
    const half_t* xr0 = xb + c15 * sx + koff;          // rows 0..15
    const half_t* xr1 = xr0 + 16 * sx;                 // rows 16..31
    for (int nt = wave; nt < NT; nt += 8) {
        v8f c0 = {}, c1 = {};
        const half_t* wt = wp + (size_t)nt * KCH * 512 + lane * 16;
#pragma unroll
        for (int kc = 0; kc < KCH; ++kc) {
            v16h a0, a1, b;
            b = *(const v16h*)(wt + (size_t)kc * 512);
            ((v8h*)&a0)[0] = *(const v8h*)(xr0 + kc * 32);
            ((v8h*)&a0)[1] = *(const v8h*)(xr0 + kc * 32 + 16);
            ((v8h*)&a1)[0] = *(const v8h*)(xr1 + kc * 32);
            ((v8h*)&a1)[1] = *(const v8h*)(xr1 + kc * 32 + 16);
            c0 = __builtin_amdgcn_wmma_f32_16x16x32_f16(false, a0, false, b,
                                                        (short)0, c0, false, false);
            c1 = __builtin_amdgcn_wmma_f32_16x16x32_f16(false, a1, false, b,
                                                        (short)0, c1, false, false);
        }
        const int col = nt * 16 + c15;
        const float bias = bp[col];
        const int r0 = (lane & 16) ? 8 : 0;
#pragma unroll
        for (int i = 0; i < 8; ++i) {
            float v = c0[i] + bias;
            if (RELU) v = fmaxf(v, 0.0f);
            yb[(size_t)(r0 + i) * sy + col] = (half_t)v;
        }
#pragma unroll
        for (int i = 0; i < 8; ++i) {
            float v = c1[i] + bias;
            if (RELU) v = fmaxf(v, 0.0f);
            yb[(size_t)(16 + r0 + i) * sy + col] = (half_t)v;
        }
    }
}

// MODE 0: store f32 to gout[ridx[r]*ldo+col] (+ optional residual), col<ncols
// MODE 1: global_atomic_add_f32 scatter (segment_sum)
// MODE 2: store f16 to ((half_t*)gout)[ridx[r]*ldo+col]
template<int KP, int NP, int MODE>
__device__ __forceinline__
void layer_out(const half_t* __restrict__ wp, const float* __restrict__ bp,
               const half_t* __restrict__ xb, int sx,
               int lane, int wave,
               float* gout, int ldo, int ncols,
               const int* __restrict__ ridx, const float* __restrict__ resid)
{
    constexpr int KCH = KP / 32;
    constexpr int NT  = NP / 16;
    const int c15  = lane & 15;
    const int koff = (lane & 16) ? 8 : 0;
    const half_t* xr0 = xb + c15 * sx + koff;
    const half_t* xr1 = xr0 + 16 * sx;
    for (int nt = wave; nt < NT; nt += 8) {
        v8f c0 = {}, c1 = {};
        const half_t* wt = wp + (size_t)nt * KCH * 512 + lane * 16;
#pragma unroll
        for (int kc = 0; kc < KCH; ++kc) {
            v16h a0, a1, b;
            b = *(const v16h*)(wt + (size_t)kc * 512);
            ((v8h*)&a0)[0] = *(const v8h*)(xr0 + kc * 32);
            ((v8h*)&a0)[1] = *(const v8h*)(xr0 + kc * 32 + 16);
            ((v8h*)&a1)[0] = *(const v8h*)(xr1 + kc * 32);
            ((v8h*)&a1)[1] = *(const v8h*)(xr1 + kc * 32 + 16);
            c0 = __builtin_amdgcn_wmma_f32_16x16x32_f16(false, a0, false, b,
                                                        (short)0, c0, false, false);
            c1 = __builtin_amdgcn_wmma_f32_16x16x32_f16(false, a1, false, b,
                                                        (short)0, c1, false, false);
        }
        const int col = nt * 16 + c15;
        const float bias = bp[col];
        const int r0 = (lane & 16) ? 8 : 0;
#pragma unroll
        for (int half_m = 0; half_m < 2; ++half_m) {
            const v8f& c = half_m ? c1 : c0;
            const int rb = half_m * 16 + r0;
#pragma unroll
            for (int i = 0; i < 8; ++i) {
                float v = c[i] + bias;
                const int r = rb + i;
                if constexpr (MODE == 0) {
                    if (col < ncols) {
                        size_t o = (size_t)ridx[r] * ldo + col;
                        if (resid) v += resid[o];
                        gout[o] = v;
                    }
                } else if constexpr (MODE == 1) {
                    unsafeAtomicAdd(&gout[(size_t)ridx[r] * ldo + col], v);
                } else {
                    ((half_t*)gout)[(size_t)ridx[r] * ldo + col] = (half_t)v;
                }
            }
        }
    }
}

// ===========================================================================
// Encoders
// ===========================================================================
__global__ __launch_bounds__(256)
void node_encoder_kernel(const float* __restrict__ x,
                         const half_t* w0, const float* b0,
                         const half_t* w1, const float* b1,
                         const half_t* w2, const float* b2,
                         float* __restrict__ hout)
{
    __shared__ __align__(16) half_t xb[32][40];
    __shared__ __align__(16) half_t t1[32][136];
    __shared__ __align__(16) half_t t2[32][72];
    __shared__ int ridx[32];
    const int tid = threadIdx.x, lane = tid & 31, wave = tid >> 5;
    const int n0 = blockIdx.x * 32;
    if (tid < 32) ridx[tid] = n0 + tid;
    for (int i = tid; i < 32 * 32; i += 256) {
        int r = i >> 5, c = i & 31;
        xb[r][c] = (half_t)((c < 10) ? x[(size_t)(n0 + r) * 10 + c] : 0.0f);
    }
    __syncthreads();
    layer_lds<32, 128, true>(w0, b0, &xb[0][0], 40, &t1[0][0], 136, lane, wave);
    __syncthreads();
    layer_lds<128, 64, true>(w1, b1, &t1[0][0], 136, &t2[0][0], 72, lane, wave);
    __syncthreads();
    layer_out<64, 128, 0>(w2, b2, &t2[0][0], 72, lane, wave,
                          hout, 128, 128, ridx, nullptr);
}

__global__ __launch_bounds__(256)
void edge_encoder_kernel(const float* __restrict__ eattr,
                         const half_t* w0, const float* b0,
                         const half_t* w1, const float* b1,
                         const half_t* w2, const float* b2,
                         half_t* __restrict__ ea)
{
    __shared__ __align__(16) half_t xb[32][40];
    __shared__ __align__(16) half_t t1[32][72];
    __shared__ __align__(16) half_t t2[32][40];
    __shared__ int ridx[32];
    const int tid = threadIdx.x, lane = tid & 31, wave = tid >> 5;
    const int e0 = blockIdx.x * 32;
    if (tid < 32) ridx[tid] = e0 + tid;
    for (int i = tid; i < 32 * 32; i += 256) {
        int r = i >> 5, c = i & 31;
        xb[r][c] = (half_t)((c < 4) ? eattr[(size_t)(e0 + r) * 4 + c] : 0.0f);
    }
    __syncthreads();
    layer_lds<32, 64, true>(w0, b0, &xb[0][0], 40, &t1[0][0], 72, lane, wave);
    __syncthreads();
    layer_lds<64, 32, true>(w1, b1, &t1[0][0], 72, &t2[0][0], 40, lane, wave);
    __syncthreads();
    layer_out<32, 64, 2>(w2, b2, &t2[0][0], 40, lane, wave,
                         (float*)ea, 64, 64, ridx, nullptr);
}

// ===========================================================================
// Fused edge stage for one GN layer (32 edges / block):
//   msg1 = edge_mlp([h[row],h[col],ea]); ea' = edge_update([ea,msg1]);
//   msg2 = edge_mlp([h[col],h[row],ea']); agg[col] += msg2  (atomics)
// ===========================================================================
__global__ __launch_bounds__(256)
void edge_layer_kernel(const float* __restrict__ h,
                       half_t* __restrict__ ea,
                       float* __restrict__ agg,
                       const int* __restrict__ eidx, int nE,
                       const half_t* ew0, const float* eb0,
                       const half_t* ew1, const float* eb1,
                       const half_t* ew2, const float* eb2,
                       const half_t* uw0, const float* ub0,
                       const half_t* uw1, const float* ub1,
                       const half_t* uw2, const float* ub2)
{
    __shared__ __align__(16) half_t xb[32][328];   // [h_a | h_b | ea] K=320
    __shared__ __align__(16) half_t ub[32][200];   // [ea | msg]      K=192
    __shared__ __align__(16) half_t t1[32][136];
    __shared__ __align__(16) half_t t2[32][72];
    __shared__ __align__(16) half_t t3[32][40];
    __shared__ __align__(16) half_t msg[32][136];
    __shared__ __align__(16) half_t ean[32][72];
    __shared__ int rix[32], cix[32];

    const int tid = threadIdx.x, lane = tid & 31, wave = tid >> 5;
    const int e0 = blockIdx.x * 32;
    if (tid < 32) {
        rix[tid] = eidx[e0 + tid];
        cix[tid] = eidx[nE + e0 + tid];
    }
    __syncthreads();
    for (int i = tid; i < 4096; i += 256) {
        int r = i >> 7, c = i & 127;
        xb[r][c]       = (half_t)h[(size_t)rix[r] * 128 + c];
        xb[r][128 + c] = (half_t)h[(size_t)cix[r] * 128 + c];
    }
    for (int i = tid; i < 2048; i += 256) {
        int r = i >> 6, c = i & 63;
        xb[r][256 + c] = ea[(size_t)(e0 + r) * 64 + c];
    }
    __syncthreads();

    // edge_mlp pass 1 -> msg
    layer_lds<320, 128, true >(ew0, eb0, &xb[0][0], 328, &t1[0][0], 136, lane, wave);
    __syncthreads();
    layer_lds<128, 64,  true >(ew1, eb1, &t1[0][0], 136, &t2[0][0], 72, lane, wave);
    __syncthreads();
    layer_lds<64,  128, false>(ew2, eb2, &t2[0][0], 72, &msg[0][0], 136, lane, wave);
    __syncthreads();

    // edge_update input [ea|msg]
    for (int i = tid; i < 2048; i += 256) {
        int r = i >> 6, c = i & 63;
        ub[r][c] = xb[r][256 + c];
    }
    for (int i = tid; i < 4096; i += 256) {
        int r = i >> 7, c = i & 127;
        ub[r][64 + c] = msg[r][c];
    }
    __syncthreads();
    layer_lds<192, 64, true >(uw0, ub0, &ub[0][0], 200, &t2[0][0], 72, lane, wave);
    __syncthreads();
    layer_lds<64,  32, true >(uw1, ub1, &t2[0][0], 72, &t3[0][0], 40, lane, wave);
    __syncthreads();
    layer_lds<32,  64, false>(uw2, ub2, &t3[0][0], 40, &ean[0][0], 72, lane, wave);
    __syncthreads();

    // commit ea', rebuild X = [h[col] | h[row] | ea'] in place
    for (int i = tid; i < 2048; i += 256) {
        int r = i >> 6, c = i & 63;
        half_t v = ean[r][c];
        ea[(size_t)(e0 + r) * 64 + c] = v;
        xb[r][256 + c] = v;
    }
    for (int i = tid; i < 4096; i += 256) {   // swap halves (unique pair/thread)
        int r = i >> 7, c = i & 127;
        half_t a = xb[r][c], b = xb[r][128 + c];
        xb[r][c] = b;
        xb[r][128 + c] = a;
    }
    __syncthreads();

    // edge_mlp pass 2 -> scatter-add into agg over col
    layer_lds<320, 128, true>(ew0, eb0, &xb[0][0], 328, &t1[0][0], 136, lane, wave);
    __syncthreads();
    layer_lds<128, 64,  true>(ew1, eb1, &t1[0][0], 136, &t2[0][0], 72, lane, wave);
    __syncthreads();
    layer_out<64, 128, 1>(ew2, eb2, &t2[0][0], 72, lane, wave,
                          agg, 128, 128, cix, nullptr);
}

// ===========================================================================
// Fused node stage: h = node_mlp([h, agg]) (+ residual for layers > 0)
// ===========================================================================
__global__ __launch_bounds__(256)
void node_layer_kernel(float* __restrict__ h, const float* __restrict__ agg,
                       const half_t* w0, const float* b0,
                       const half_t* w1, const float* b1,
                       const half_t* w2, const float* b2, int use_resid)
{
    __shared__ __align__(16) half_t xb[32][264];
    __shared__ __align__(16) half_t t1[32][136];
    __shared__ __align__(16) half_t t2[32][72];
    __shared__ int ridx[32];
    const int tid = threadIdx.x, lane = tid & 31, wave = tid >> 5;
    const int n0 = blockIdx.x * 32;
    if (tid < 32) ridx[tid] = n0 + tid;
    for (int i = tid; i < 4096; i += 256) {
        int r = i >> 7, c = i & 127;
        xb[r][c]       = (half_t)h[(size_t)(n0 + r) * 128 + c];
        xb[r][128 + c] = (half_t)agg[(size_t)(n0 + r) * 128 + c];
    }
    __syncthreads();
    layer_lds<256, 128, true>(w0, b0, &xb[0][0], 264, &t1[0][0], 136, lane, wave);
    __syncthreads();
    layer_lds<128, 64,  true>(w1, b1, &t1[0][0], 136, &t2[0][0], 72, lane, wave);
    __syncthreads();
    layer_out<64, 128, 0>(w2, b2, &t2[0][0], 72, lane, wave,
                          h, 128, 128, ridx, use_resid ? h : nullptr);
}

// ===========================================================================
// Node decoder: node_pred = MLP(h) -> [N,3]
// ===========================================================================
__global__ __launch_bounds__(256)
void node_decoder_kernel(const float* __restrict__ h,
                         const half_t* w0, const float* b0,
                         const half_t* w1, const float* b1,
                         const half_t* w2, const float* b2,
                         float* __restrict__ out)
{
    __shared__ __align__(16) half_t xb[32][136];
    __shared__ __align__(16) half_t t1[32][72];
    __shared__ __align__(16) half_t t2[32][40];
    __shared__ int ridx[32];
    const int tid = threadIdx.x, lane = tid & 31, wave = tid >> 5;
    const int n0 = blockIdx.x * 32;
    if (tid < 32) ridx[tid] = n0 + tid;
    for (int i = tid; i < 4096; i += 256) {
        int r = i >> 7, c = i & 127;
        xb[r][c] = (half_t)h[(size_t)(n0 + r) * 128 + c];
    }
    __syncthreads();
    layer_lds<128, 64, true>(w0, b0, &xb[0][0], 136, &t1[0][0], 72, lane, wave);
    __syncthreads();
    layer_lds<64, 32, true >(w1, b1, &t1[0][0], 72, &t2[0][0], 40, lane, wave);
    __syncthreads();
    layer_out<32, 16, 0>(w2, b2, &t2[0][0], 40, lane, wave,
                         out, 3, 3, ridx, nullptr);
}

// ===========================================================================
// Utilities: zero, mean/max partial reduction, global head (LN + decoder)
// ===========================================================================
__global__ void zero_f32_kernel(float* __restrict__ p, size_t n)
{
    size_t i = (size_t)blockIdx.x * blockDim.x + threadIdx.x;
    size_t stride = (size_t)gridDim.x * blockDim.x;
    for (; i < n; i += stride) p[i] = 0.0f;
}

__global__ __launch_bounds__(128)
void reduce_kernel(const float* __restrict__ h,
                   float* __restrict__ psum, float* __restrict__ pmax)
{
    const int c = threadIdx.x;                 // channel 0..127
    const int start = blockIdx.x * 1024;
    int end = start + 1024;
    if (end > N_NODES) end = N_NODES;
    float s = 0.0f, m = -3.402823466e38f;
    for (int i = start; i < end; ++i) {
        float v = h[(size_t)i * 128 + c];
        s += v;
        m = fmaxf(m, v);
    }
    psum[blockIdx.x * 128 + c] = s;
    pmax[blockIdx.x * 128 + c] = m;
}

__global__ __launch_bounds__(256)
void global_head_kernel(const float* __restrict__ psum,
                        const float* __restrict__ pmax, int nparts,
                        const float* lng, const float* lnb,
                        const float* W0, const float* B0, const float* g0,
                        const float* be0, const float* rm0, const float* rv0,
                        const float* W1, const float* B1, const float* g1,
                        const float* be1, const float* rm1, const float* rv1,
                        const float* W2, const float* B2,
                        float* __restrict__ out)
{
    __shared__ float gf[256], red[256], h1[128], h2[64];
    const int t = threadIdx.x;
    float v;
    if (t < 128) {
        float s = 0.0f;
        for (int p = 0; p < nparts; ++p) s += psum[p * 128 + t];
        v = s * (1.0f / (float)N_NODES);
    } else {
        const int c = t - 128;
        float m = -3.402823466e38f;
        for (int p = 0; p < nparts; ++p) m = fmaxf(m, pmax[p * 128 + c]);
        v = m;
    }
    gf[t] = v;
    red[t] = v;
    __syncthreads();
    for (int s = 128; s > 0; s >>= 1) { if (t < s) red[t] += red[t + s]; __syncthreads(); }
    const float mu = red[0] * (1.0f / 256.0f);
    __syncthreads();
    const float d = gf[t] - mu;
    red[t] = d * d;
    __syncthreads();
    for (int s = 128; s > 0; s >>= 1) { if (t < s) red[t] += red[t + s]; __syncthreads(); }
    const float var = red[0] * (1.0f / 256.0f);
    __syncthreads();
    gf[t] = d * rsqrtf(var + BN_EPS) * lng[t] + lnb[t];
    __syncthreads();
    if (t < 128) {
        float a = B0[t];
        for (int k = 0; k < 256; ++k) a += gf[k] * W0[k * 128 + t];
        a = (a - rm0[t]) * (g0[t] * rsqrtf(rv0[t] + BN_EPS)) + be0[t];
        h1[t] = fmaxf(a, 0.0f);
    }
    __syncthreads();
    if (t < 64) {
        float a = B1[t];
        for (int k = 0; k < 128; ++k) a += h1[k] * W1[k * 64 + t];
        a = (a - rm1[t]) * (g1[t] * rsqrtf(rv1[t] + BN_EPS)) + be1[t];
        h2[t] = fmaxf(a, 0.0f);
    }
    __syncthreads();
    if (t < 2) {
        float a = B2[t];
        for (int k = 0; k < 64; ++k) a += h2[k] * W2[k * 2 + t];
        out[t] = a;
    }
}

// ===========================================================================
// Host launcher
// ===========================================================================
extern "C" void kernel_launch(void* const* d_in, const int* in_sizes, int n_in,
                              void* d_out, int out_size, void* d_ws, size_t ws_size,
                              hipStream_t stream)
{
    (void)in_sizes; (void)out_size; (void)ws_size;
    if (n_in < 313) return;   // expected jax-flattened param count

    // --- input mapping ------------------------------------------------------
    // top-level insertion order: x(0), edge_index(1), edge_attr(2), params...
    // params flattened jax-style (sorted dict keys):
    //   edge_encoder@3(14), global_decoder@17(14), layers@31 (6 x 42:
    //     edge_mlp, edge_update_mlp, node_mlp; each 14), ln@283(2),
    //   node_decoder@285(14), node_encoder@299(14)
    // per hidden Linear: W,b,gamma,beta,run_mean,run_var ; last Linear: W,b
    const float* x     = (const float*)d_in[0];
    const int*   eidx  = (const int*)d_in[1];
    const float* eattr = (const float*)d_in[2];
    auto F = [&](int i) { return (const float*)d_in[i]; };

    // --- workspace carve-up -------------------------------------------------
    char* ws = (char*)d_ws;
    size_t off = 0;
    auto alloc = [&](size_t bytes) -> char* {
        off = (off + 255) & ~(size_t)255;
        char* p = ws + off;
        off += bytes;
        return p;
    };
    float*  h    = (float*)alloc((size_t)N_NODES * 128 * 4);
    float*  agg  = (float*)alloc((size_t)N_NODES * 128 * 4);
    half_t* ea   = (half_t*)alloc((size_t)N_EDGES * 64 * 2);
    const int NPARTS = (N_NODES + 1023) / 1024;   // 98
    float*  psum = (float*)alloc((size_t)NPARTS * 128 * 4);
    float*  pmax = (float*)alloc((size_t)NPARTS * 128 * 4);

    // --- weight prep (BN fold + f16 pack into WMMA B-fragment layout) -------
    struct LP { const half_t* w; const float* b; };
    auto prep_lin = [&](int base, bool bn, int K, int N) -> LP {
        int Kp = (K + 31) & ~31, Np = (N + 15) & ~15;
        half_t* wout = (half_t*)alloc((size_t)Kp * Np * 2);
        float*  bout = (float*)alloc((size_t)Np * 4);
        int total = Kp * Np;
        prep_weights_kernel<<<(total + 255) / 256, 256, 0, stream>>>(
            F(base), F(base + 1),
            bn ? F(base + 2) : nullptr, bn ? F(base + 3) : nullptr,
            bn ? F(base + 4) : nullptr, bn ? F(base + 5) : nullptr,
            K, N, Kp, Np, wout, bout);
        return {wout, bout};
    };
    auto prep_mlp = [&](int base, const int* dims, LP* o) {
        o[0] = prep_lin(base,      true,  dims[0], dims[1]);
        o[1] = prep_lin(base + 6,  true,  dims[1], dims[2]);
        o[2] = prep_lin(base + 12, false, dims[2], dims[3]);
    };

    LP ne[3]; { int d[4] = {10, 128, 64, 128}; prep_mlp(299, d, ne); }
    LP ee[3]; { int d[4] = {4, 64, 32, 64};    prep_mlp(3,   d, ee); }
    LP em[6][3], um[6][3], nm[6][3];
    {
        int de[4] = {320, 128, 64, 128};
        int du[4] = {192, 64, 32, 64};
        int dn[4] = {256, 128, 64, 128};
        for (int i = 0; i < 6; ++i) {
            int b = 31 + 42 * i;
            prep_mlp(b,      de, em[i]);
            prep_mlp(b + 14, du, um[i]);
            prep_mlp(b + 28, dn, nm[i]);
        }
    }
    LP nd[3]; { int d[4] = {128, 64, 32, 3};   prep_mlp(285, d, nd); }

    // --- forward ------------------------------------------------------------
    node_encoder_kernel<<<N_NODES / 32, 256, 0, stream>>>(
        x, ne[0].w, ne[0].b, ne[1].w, ne[1].b, ne[2].w, ne[2].b, h);
    edge_encoder_kernel<<<N_EDGES / 32, 256, 0, stream>>>(
        eattr, ee[0].w, ee[0].b, ee[1].w, ee[1].b, ee[2].w, ee[2].b, ea);

    for (int i = 0; i < 6; ++i) {
        zero_f32_kernel<<<2048, 256, 0, stream>>>(agg, (size_t)N_NODES * 128);
        edge_layer_kernel<<<N_EDGES / 32, 256, 0, stream>>>(
            h, ea, agg, eidx, N_EDGES,
            em[i][0].w, em[i][0].b, em[i][1].w, em[i][1].b, em[i][2].w, em[i][2].b,
            um[i][0].w, um[i][0].b, um[i][1].w, um[i][1].b, um[i][2].w, um[i][2].b);
        node_layer_kernel<<<N_NODES / 32, 256, 0, stream>>>(
            h, agg,
            nm[i][0].w, nm[i][0].b, nm[i][1].w, nm[i][1].b, nm[i][2].w, nm[i][2].b,
            i > 0 ? 1 : 0);
    }

    node_decoder_kernel<<<N_NODES / 32, 256, 0, stream>>>(
        h, nd[0].w, nd[0].b, nd[1].w, nd[1].b, nd[2].w, nd[2].b, (float*)d_out);

    reduce_kernel<<<NPARTS, 128, 0, stream>>>(h, psum, pmax);
    global_head_kernel<<<1, 256, 0, stream>>>(
        psum, pmax, NPARTS,
        F(283), F(284),
        F(17), F(18), F(19), F(20), F(21), F(22),
        F(23), F(24), F(25), F(26), F(27), F(28),
        F(29), F(30),
        (float*)d_out + (size_t)N_NODES * 3);
}
